// PushPull_Attention_64407329571106
// MI455X (gfx1250) — compile-verified, matrix operand order checked by reading, not run on hardware
//
#include <hip/hip_runtime.h>

#define T_DIM 4
#define B_DIM 16
#define C_DIM 256
#define N_DIM 1024
#define NH 8
#define CH 32
#define TBATCH (T_DIM * B_DIM)          // 64
#define EPSV 1e-5f
#define TBCN ((size_t)TBATCH * C_DIM * N_DIM)   // 16,777,216 elements

typedef __bf16 bf16_t;
typedef __attribute__((ext_vector_type(16))) __bf16 v16bf;
typedef __attribute__((ext_vector_type(8)))  __bf16 v8bf;
typedef __attribute__((ext_vector_type(4)))  __bf16 v4bf;
typedef __attribute__((ext_vector_type(8)))  float  v8f;
typedef __attribute__((ext_vector_type(4)))  int    v4i_t;

#define AS1 __attribute__((address_space(1)))
#define AS3 __attribute__((address_space(3)))

#if defined(__gfx1250__) && __has_builtin(__builtin_amdgcn_global_load_async_to_lds_b128)
#define USE_ASYNC_LDS 1
#else
#define USE_ASYNC_LDS 0
#endif

#if USE_ASYNC_LDS
#if __has_builtin(__builtin_amdgcn_s_wait_asynccnt)
#define WAIT_ASYNC() __builtin_amdgcn_s_wait_asynccnt(0)
#else
#define WAIT_ASYNC() asm volatile("s_wait_asynccnt 0" ::: "memory")
#endif
#endif

// ---------------------------------------------------------------------------
// Fold eval-mode BN into weights: W'[o,c] = s[o]*W[o,c]; bias per branch.
// sel 0: q, 1: k (bias = beta - s*mean), 2: proj (bias = s*(proj_b-mean)+beta)
// ---------------------------------------------------------------------------
__global__ __launch_bounds__(256) void fold_bn_kernel(
    const float* __restrict__ qw, const float* __restrict__ qg,
    const float* __restrict__ qb, const float* __restrict__ qm,
    const float* __restrict__ qv,
    const float* __restrict__ kw, const float* __restrict__ kg,
    const float* __restrict__ kb, const float* __restrict__ km,
    const float* __restrict__ kv,
    const float* __restrict__ pw, const float* __restrict__ pbias,
    const float* __restrict__ pg, const float* __restrict__ pbeta,
    const float* __restrict__ pm, const float* __restrict__ pv,
    bf16_t* __restrict__ w_all, float* __restrict__ bias_all) {
  int id  = blockIdx.x * 256 + threadIdx.x;   // 0 .. 3*65536-1
  int sel = id >> 16;
  int e   = id & 65535;
  int o   = e >> 8;
  int c   = e & 255;
  const float* W;
  float g, b, m, v, extra = 0.0f;
  if (sel == 0)      { W = qw; g = qg[o]; b = qb[o];    m = qm[o]; v = qv[o]; }
  else if (sel == 1) { W = kw; g = kg[o]; b = kb[o];    m = km[o]; v = kv[o]; }
  else               { W = pw; g = pg[o]; b = pbeta[o]; m = pm[o]; v = pv[o];
                       extra = pbias[o]; }
  float s = g * rsqrtf(v + EPSV);
  w_all[id] = (bf16_t)(W[e] * s);
  if (c == 0) {
    bias_all[sel * 256 + o] = (sel == 2) ? (s * (extra - m) + b) : (b - s * m);
  }
}

// ---------------------------------------------------------------------------
// x fp32 -> bf16 (layout unchanged: [T,B,C,N])
// ---------------------------------------------------------------------------
__global__ __launch_bounds__(256) void cvt_x_kernel(const float* __restrict__ x,
                                                    bf16_t* __restrict__ xb) {
  size_t i = ((size_t)blockIdx.x * 256 + threadIdx.x) * 4;
  float4 f = *(const float4*)(x + i);
  v4bf o;
  o[0] = (bf16_t)f.x; o[1] = (bf16_t)f.y; o[2] = (bf16_t)f.z; o[3] = (bf16_t)f.w;
  *(v4bf*)(xb + i) = o;
}

// ---------------------------------------------------------------------------
// Batched WMMA GEMM: Y[sel][tb] = W[sel] @ X[tb] + bias[sel]   (bf16 in/out,
// f32 accumulate). Block tile 128x128, 8 waves (2x4), wave tile 64x32,
// K-steps of 32 using v_wmma_f32_16x16x32_bf16.
// nmat==2: blockIdx.z = tb*2+sel (q & k share X). nmat==1: blockIdx.z = tb.
// ---------------------------------------------------------------------------
__global__ __launch_bounds__(256) void wmma_gemm_kernel(
    const bf16_t* __restrict__ Wall, const float* __restrict__ Ball,
    const bf16_t* __restrict__ X, bf16_t* __restrict__ Y, int nmat) {
  __shared__ __align__(16) bf16_t ldsA[128 * 32];   // [m][k]
  __shared__ __align__(16) bf16_t ldsB[128 * 32];   // [n][k]

  const int tid  = threadIdx.x;
  const int lane = tid & 31;
  const int wave = tid >> 5;
  const int wm   = wave >> 2;     // 0..1
  const int wn   = wave & 3;      // 0..3

  int z   = blockIdx.z;
  int sel = (nmat == 2) ? (z & 1)  : 0;
  int tb  = (nmat == 2) ? (z >> 1) : z;

  const bf16_t* Wm = Wall + (size_t)sel * C_DIM * C_DIM;
  const float*  bs = Ball + sel * C_DIM;
  const bf16_t* Xb = X + (size_t)tb * C_DIM * N_DIM;
  bf16_t*       Yb = Y + ((size_t)sel * TBATCH + tb) * (size_t)(C_DIM * N_DIM);

  const int m0 = blockIdx.y * 128;
  const int n0 = blockIdx.x * 128;

  v8f acc[4][2];
#pragma unroll
  for (int i = 0; i < 4; ++i)
#pragma unroll
    for (int j = 0; j < 2; ++j)
#pragma unroll
      for (int g = 0; g < 8; ++g) acc[i][j][g] = 0.0f;

  for (int k0 = 0; k0 < C_DIM; k0 += 32) {
    // ---- A tile: W rows [m0,m0+128) x K cols [k0,k0+32), row-major in LDS --
#pragma unroll
    for (int i = 0; i < 2; ++i) {
      int cid = tid + i * 256;          // 0..511 : 512 chunks of 16B
      int r   = cid >> 2;               // 0..127
      int cc  = (cid & 3) * 8;          // 0,8,16,24
      const bf16_t* gp = Wm + (size_t)(m0 + r) * C_DIM + (k0 + cc);
      bf16_t*       lp = ldsA + r * 32 + cc;
#if USE_ASYNC_LDS
      __builtin_amdgcn_global_load_async_to_lds_b128((AS1 v4i_t*)gp, (AS3 v4i_t*)lp, 0, 0);
#else
      *(uint4*)lp = *(const uint4*)gp;
#endif
    }
    // ---- B tile: X rows [k0,k0+32) x N cols [n0,n0+128), transposed to [n][k]
    {
      int kp   = tid & 15;              // k pair index -> k = 2*kp
      int gidx = tid >> 4;              // column group  -> n = gidx*8..+7
      const bf16_t* gp0 = Xb + (size_t)(k0 + kp * 2) * N_DIM + (n0 + gidx * 8);
      const bf16_t* gp1 = gp0 + N_DIM;
      v8bf r0 = *(const v8bf*)gp0;
      v8bf r1 = *(const v8bf*)gp1;
#pragma unroll
      for (int u = 0; u < 8; ++u) {
        bf16_t* lp = ldsB + (gidx * 8 + u) * 32 + kp * 2;
        lp[0] = r0[u];
        lp[1] = r1[u];
      }
    }
#if USE_ASYNC_LDS
    WAIT_ASYNC();
#endif
    __syncthreads();

    // ---- fragment loads (16-bit WMMA operand layout: per-lane K chunks
    //      {kb..kb+7} and {kb+16..kb+23}, kb = (lane>>4)*8) -----------------
    const int kb = (lane >> 4) * 8;
    v16bf afrag[4], bfrag[2];
#pragma unroll
    for (int i = 0; i < 4; ++i) {
      int row = wm * 64 + i * 16 + (lane & 15);
      v8bf lo = *(const v8bf*)(ldsA + row * 32 + kb);
      v8bf hi = *(const v8bf*)(ldsA + row * 32 + kb + 16);
      afrag[i] = __builtin_shufflevector(lo, hi, 0, 1, 2, 3, 4, 5, 6, 7,
                                         8, 9, 10, 11, 12, 13, 14, 15);
    }
#pragma unroll
    for (int j = 0; j < 2; ++j) {
      int col = wn * 32 + j * 16 + (lane & 15);
      v8bf lo = *(const v8bf*)(ldsB + col * 32 + kb);
      v8bf hi = *(const v8bf*)(ldsB + col * 32 + kb + 16);
      bfrag[j] = __builtin_shufflevector(lo, hi, 0, 1, 2, 3, 4, 5, 6, 7,
                                         8, 9, 10, 11, 12, 13, 14, 15);
    }
#pragma unroll
    for (int i = 0; i < 4; ++i)
#pragma unroll
      for (int j = 0; j < 2; ++j)
        acc[i][j] = __builtin_amdgcn_wmma_f32_16x16x32_bf16(
            false, afrag[i], false, bfrag[j], (short)0, acc[i][j], false, false);

    __syncthreads();
  }

  // ---- epilogue: bias add, bf16 store. C/D layout: VGPR g -> M = g + 8*(lane>>4),
  //      N = lane&15 ------------------------------------------------------
#pragma unroll
  for (int i = 0; i < 4; ++i) {
    int mbase = m0 + wm * 64 + i * 16 + (lane >> 4) * 8;
#pragma unroll
    for (int j = 0; j < 2; ++j) {
      int ncol = n0 + wn * 32 + j * 16 + (lane & 15);
#pragma unroll
      for (int g = 0; g < 8; ++g) {
        int row = mbase + g;
        float val = acc[i][j][g] + bs[row];
        Yb[(size_t)row * N_DIM + ncol] = (bf16_t)val;
      }
    }
  }
}

// ---------------------------------------------------------------------------
// LIF + push-pull attention. One wave32 per (b, head, n); lane = head-channel.
// States kept in registers over T; CH-sum via wave-wide shuffle reduction.
// x_one (= attn * k_spike, exactly {0,1}) written as bf16.
// ---------------------------------------------------------------------------
__global__ __launch_bounds__(256) void lif_attn_kernel(
    const bf16_t* __restrict__ qbuf, const bf16_t* __restrict__ kbuf,
    bf16_t* __restrict__ xone) {
  int tid  = blockIdx.x * 256 + threadIdx.x;
  int lane = threadIdx.x & 31;
  int wg   = tid >> 5;          // 0 .. B*NH*N-1 = 131071
  int b    = wg >> 13;          // / (NH*N)
  int rem  = wg & 8191;
  int hh   = rem >> 10;
  int n    = rem & 1023;
  int ch   = hh * CH + lane;

  float vqe = 0.0f, vqi = 0.0f, vk = 0.0f, va = 0.0f;
#pragma unroll
  for (int t = 0; t < T_DIM; ++t) {
    size_t idx = ((((size_t)t * B_DIM + b) * C_DIM + ch) << 10) + n;
    float qv = (float)qbuf[idx];
    float kv = (float)kbuf[idx];
    // tau = 2.0 -> charge h = 0.5*(v + x); hard reset, detached
    float he = 0.5f * (vqe + qv); float se = (he >= 1.0f) ? 1.0f : 0.0f; vqe = he * (1.0f - se);
    float hi = 0.5f * (vqi - qv); float si = (hi >= 1.0f) ? 1.0f : 0.0f; vqi = hi * (1.0f - si);
    float hk = 0.5f * (vk  + kv); float sk = (hk >= 1.0f) ? 1.0f : 0.0f; vk  = hk * (1.0f - sk);
    float d = se - si;
#pragma unroll
    for (int off = 16; off >= 1; off >>= 1) d += __shfl_xor(d, off, 32);
    float ha = 0.5f * (va + d); float sa = (ha >= 0.5f) ? 1.0f : 0.0f; va = ha * (1.0f - sa);
    xone[idx] = (bf16_t)(sa * sk);
  }
}

// ---------------------------------------------------------------------------
// Final LIF over projected activations -> fp32 spikes [T,B,C,H,W]
// ---------------------------------------------------------------------------
__global__ __launch_bounds__(256) void final_lif_kernel(
    const bf16_t* __restrict__ p, float* __restrict__ out) {
  size_t i = (size_t)blockIdx.x * 256 + threadIdx.x;     // 0 .. B*C*N-1
  const size_t ts = (size_t)B_DIM * C_DIM * N_DIM;       // 4,194,304
  float v = 0.0f;
#pragma unroll
  for (int t = 0; t < T_DIM; ++t) {
    float pv = (float)p[t * ts + i];
    float h  = 0.5f * (v + pv);
    float s  = (h >= 1.0f) ? 1.0f : 0.0f;
    v = h * (1.0f - s);
    out[t * ts + i] = s;
  }
}

// ---------------------------------------------------------------------------
extern "C" void kernel_launch(void* const* d_in, const int* in_sizes, int n_in,
                              void* d_out, int out_size, void* d_ws, size_t ws_size,
                              hipStream_t stream) {
  const float* x       = (const float*)d_in[0];
  const float* q_w     = (const float*)d_in[1];
  const float* q_gamma = (const float*)d_in[2];
  const float* q_beta  = (const float*)d_in[3];
  const float* q_mean  = (const float*)d_in[4];
  const float* q_var   = (const float*)d_in[5];
  const float* k_w     = (const float*)d_in[6];
  const float* k_gamma = (const float*)d_in[7];
  const float* k_beta  = (const float*)d_in[8];
  const float* k_mean  = (const float*)d_in[9];
  const float* k_var   = (const float*)d_in[10];
  const float* proj_w  = (const float*)d_in[11];
  const float* proj_b  = (const float*)d_in[12];
  const float* p_gamma = (const float*)d_in[13];
  const float* p_beta  = (const float*)d_in[14];
  const float* p_mean  = (const float*)d_in[15];
  const float* p_var   = (const float*)d_in[16];
  float* out = (float*)d_out;

  uint8_t* ws = (uint8_t*)d_ws;
  // layout: [0, 384K) folded bf16 weights; [384K, ..) biases; 1MB: xbf (32MB);
  // 33MB..97MB: q|k bf16 (also reused for p). xone aliases xbf.
  bf16_t* w_all    = (bf16_t*)(ws + 0);
  float*  bias_all = (float*) (ws + 3u * C_DIM * C_DIM * 2u);           // 393216
  bf16_t* xbf      = (bf16_t*)(ws + (1u << 20));                        // 1 MB
  bf16_t* qk       = (bf16_t*)(ws + (1u << 20) + TBCN * sizeof(bf16_t)); // +32MB
  bf16_t* xone     = xbf;   // x no longer needed after first GEMM
  bf16_t* pbf      = qk;    // q/k no longer needed after lif_attn

  // 1. fold BN into bf16 weights + biases (3 * 256*256 elements)
  fold_bn_kernel<<<768, 256, 0, stream>>>(
      q_w, q_gamma, q_beta, q_mean, q_var,
      k_w, k_gamma, k_beta, k_mean, k_var,
      proj_w, proj_b, p_gamma, p_beta, p_mean, p_var,
      w_all, bias_all);

  // 2. x fp32 -> bf16 (16.7M elements, 4/thread)
  cvt_x_kernel<<<16384, 256, 0, stream>>>(x, xbf);

  // 3. q & k GEMMs: grid (N/128, C/128, TB*2)
  dim3 gqk(N_DIM / 128, C_DIM / 128, TBATCH * 2);
  wmma_gemm_kernel<<<gqk, 256, 0, stream>>>(w_all, bias_all, xbf, qk, 2);

  // 4. LIF + push-pull attention: one wave per (b,head,n)
  lif_attn_kernel<<<16384, 256, 0, stream>>>(qk, qk + TBCN, xone);

  // 5. proj GEMM
  dim3 gp(N_DIM / 128, C_DIM / 128, TBATCH);
  wmma_gemm_kernel<<<gp, 256, 0, stream>>>(w_all + 2 * C_DIM * C_DIM,
                                           bias_all + 2 * C_DIM, xone, pbf, 1);

  // 6. final LIF -> fp32 output
  final_lif_kernel<<<16384, 256, 0, stream>>>(pbf, out);
}
